// GPT_15994458210440
// MI455X (gfx1250) — matrix operand order repaired and checked by reading, not compiled, not run on hardware
//
#include <hip/hip_runtime.h>
#include <hip/hip_bf16.h>

// ---------------------------------------------------------------------------
// GPT-2 (124M) forward on gfx1250.
//  - Weights converted fp32->bf16 (pre-transposed to [N][K]) once per call.
//  - Activations between GEMMs kept bf16; residual stream fp32.
//  - GEMM: 64x64x64 tiles, double-buffered LDS, global_load_async_to_lds_b128
//    staging (ASYNCcnt), v_wmma_f32_16x16x32_bf16 compute.
//  - Attention: flash-style, async-staged Q/K, online softmax, WMMA for
//    Q@K^T and P@V.
// ---------------------------------------------------------------------------

typedef __attribute__((ext_vector_type(16))) __bf16       v16bf;
typedef __attribute__((ext_vector_type(8)))  float        v8f;
typedef __attribute__((ext_vector_type(4)))  unsigned int u32x4;

#define GPT_B   2
#define GPT_T   1024
#define GPT_L   12
#define GPT_D   768
#define GPT_FF  3072
#define GPT_V   50257
#define GPT_VP  50304            // V padded to multiple of 64
#define GPT_NH  12
#define GPT_HD  64
#define ROWS    (GPT_B * GPT_T)  // 2048

// fp32 -> bf16 round-to-nearest-even
__device__ __forceinline__ unsigned short f2bf(float f) {
  unsigned int u = __float_as_uint(f);
  u += 0x7fffu + ((u >> 16) & 1u);
  return (unsigned short)(u >> 16);
}

// Async copy 16B global -> LDS (tracked by ASYNCcnt). Lower 32 bits of a
// generic pointer to __shared__ are the wave-relative LDS offset (aperture
// truncation rule), which is what VDST of the async op wants.
__device__ __forceinline__ void async_copy_b128(const unsigned short* gsrc,
                                                unsigned short* ldst) {
  unsigned int       loff = (unsigned int)(unsigned long long)ldst;
  unsigned long long ga   = (unsigned long long)gsrc;
  asm volatile("global_load_async_to_lds_b128 %0, %1, off"
               :: "v"(loff), "v"(ga) : "memory");
}

__device__ __forceinline__ void wait_async0() {
  asm volatile("s_wait_asynccnt 0x0" ::: "memory");
}

// A-fragment (16x32 bf16): lanes 0-15 row=lane, K {0..7,16..23};
// lanes 16-31 same row, K {8..15,24..31}.
__device__ __forceinline__ v16bf load_frag_a(const unsigned short* base, int stride,
                                             int lane, int kk) {
  const int half = (lane >> 4) & 1;
  const int lr   = lane & 15;
  const unsigned short* p = base + lr * stride + kk + half * 8;
  union { v16bf v; u32x4 q[2]; } f;
  f.q[0] = *(const u32x4*)(p);
  f.q[1] = *(const u32x4*)(p + 16);
  return f.v;
}

// B-fragment (32x16 bf16) from [N][K] staging: lanes 0-15 col=lane K {0..7,16..23};
// lanes 16-31 col=lane-16 K {8..15,24..31}.
__device__ __forceinline__ v16bf load_frag_b(const unsigned short* base, int stride,
                                             int lane, int kk) {
  const int half = (lane >> 4) & 1;
  const int lr   = lane & 15;
  const unsigned short* p = base + lr * stride + kk + half * 8;
  union { v16bf v; u32x4 q[2]; } f;
  f.q[0] = *(const u32x4*)(p);
  f.q[1] = *(const u32x4*)(p + 16);
  return f.v;
}

__device__ __forceinline__ v8f wmma_bf16(v16bf a, v16bf b, v8f c) {
  return __builtin_amdgcn_wmma_f32_16x16x32_bf16(false, a, false, b, (short)0, c,
                                                 false, false);
}

// ---------------------------------------------------------------------------
// Weight conversion: fp32 [K][N] -> bf16 [N][K] (LDS-tiled transpose)
// ---------------------------------------------------------------------------
__global__ __launch_bounds__(256) void wconv_t_kernel(
    const float* __restrict__ W, unsigned short* __restrict__ Wt, int K, int N) {
  __shared__ float tile[64][65];
  const int n0 = blockIdx.x * 64, k0 = blockIdx.y * 64;
  const int r = threadIdx.x >> 2, cb = (threadIdx.x & 3) * 16;
  for (int j = 0; j < 16; ++j)
    tile[r][cb + j] = W[(size_t)(k0 + r) * N + n0 + cb + j];
  __syncthreads();
  unsigned short* out = Wt + (size_t)(n0 + r) * K + k0 + cb;
  for (int j = 0; j < 16; ++j) out[j] = f2bf(tile[cb + j][r]);
}

// wte: already [V][D]; straight copy with zero row padding to VP
__global__ __launch_bounds__(256) void wconv_pad_kernel(
    const float* __restrict__ W, unsigned short* __restrict__ Wt, int K, int N) {
  const int row = blockIdx.x;
  unsigned short* out = Wt + (size_t)row * K;
  if (row < N) {
    const float* in = W + (size_t)row * K;
    for (int j = threadIdx.x; j < K; j += 256) out[j] = f2bf(in[j]);
  } else {
    for (int j = threadIdx.x; j < K; j += 256) out[j] = 0;
  }
}

// ---------------------------------------------------------------------------
// Embedding: x[row] = wte[idx[row]] + wpe[row % T]   (fp32 residual stream)
// ---------------------------------------------------------------------------
__global__ __launch_bounds__(256) void embed_kernel(
    const float* __restrict__ wte, const float* __restrict__ wpe,
    const int* __restrict__ idx, float* __restrict__ X) {
  const int row = blockIdx.x;
  const int t   = row & (GPT_T - 1);
  const int tok = idx[row];
  const float* we = wte + (size_t)tok * GPT_D;
  const float* pe = wpe + (size_t)t * GPT_D;
  float* xo = X + (size_t)row * GPT_D;
  for (int j = threadIdx.x; j < GPT_D; j += 256) xo[j] = we[j] + pe[j];
}

// ---------------------------------------------------------------------------
// LayerNorm fp32 -> bf16: one wave per row, wave32 shuffle reduction
// ---------------------------------------------------------------------------
__global__ __launch_bounds__(256) void ln_kernel(
    const float* __restrict__ X, const float* __restrict__ g,
    const float* __restrict__ bt, unsigned short* __restrict__ Y, int rows) {
  const int wave = threadIdx.x >> 5, lane = threadIdx.x & 31;
  const int row = blockIdx.x * 8 + wave;
  if (row >= rows) return;
  const float* x = X + (size_t)row * GPT_D;
  float vals[GPT_D / 32];
  float s = 0.f, sq = 0.f;
  for (int i = 0; i < GPT_D / 32; ++i) {
    float v = x[lane + i * 32];
    vals[i] = v; s += v; sq += v * v;
  }
  for (int m = 16; m >= 1; m >>= 1) {
    s  += __shfl_xor(s, m, 32);
    sq += __shfl_xor(sq, m, 32);
  }
  const float mu  = s * (1.f / GPT_D);
  const float var = sq * (1.f / GPT_D) - mu * mu;
  const float rs  = rsqrtf(var + 1e-5f);
  unsigned short* y = Y + (size_t)row * GPT_D;
  for (int i = 0; i < GPT_D / 32; ++i) {
    int d = lane + i * 32;
    y[d] = f2bf((vals[i] - mu) * rs * g[d] + bt[d]);
  }
}

// ---------------------------------------------------------------------------
// GEMM: C[M,N] = A[M,K] @ Wt[N,K]^T, all-bf16 inputs, f32 accumulate.
// flags: 1=bias(fp32), 2=gelu, 4=residual-add(fp32 out), 16=bf16 out
// Tiles 64x64x64, double-buffered async-to-LDS staging, one barrier/stage.
// M, K multiples of 64; Wt row count padded to multiple of 64.
// ---------------------------------------------------------------------------
__global__ __launch_bounds__(256) void gemm_bf16_kernel(
    const unsigned short* __restrict__ A, const unsigned short* __restrict__ Wt,
    const float* __restrict__ bias, const float* __restrict__ Res,
    void* __restrict__ Cout, int M, int N, int K, int flags) {
  __shared__ __align__(16) unsigned short As[2][64 * 72];
  __shared__ __align__(16) unsigned short Ws[2][64 * 72];
  const int tid  = threadIdx.x;
  const int wave = tid >> 5, lane = tid & 31;
  const int half = (lane >> 4) & 1, lr = lane & 15;
  const int wr = wave & 3, wc = wave >> 2;
  const int m0 = blockIdx.y * 64, n0 = blockIdx.x * 64;
  const int r  = tid >> 2, cb = (tid & 3) * 16;  // staging: row + 16-elem chunk

  const unsigned short* arow = A  + (size_t)(m0 + r) * K + cb;
  const unsigned short* wrow = Wt + (size_t)(n0 + r) * K + cb;
  unsigned short* adst0 = &As[0][r * 72 + cb];
  unsigned short* wdst0 = &Ws[0][r * 72 + cb];
  unsigned short* adst1 = &As[1][r * 72 + cb];
  unsigned short* wdst1 = &Ws[1][r * 72 + cb];

  // prologue: stage k0 = 0 into buffer 0
  async_copy_b128(arow, adst0);
  async_copy_b128(arow + 8, adst0 + 8);
  async_copy_b128(wrow, wdst0);
  async_copy_b128(wrow + 8, wdst0 + 8);

  v8f c0 = {}, c1 = {};
  int buf = 0;
  for (int k0 = 0; k0 < K; k0 += 64) {
    wait_async0();
    __syncthreads();  // staged tile visible to all waves; prior reads done
    if (k0 + 64 < K) {  // prefetch next stage into the other buffer
      const unsigned short* an = arow + k0 + 64;
      const unsigned short* wn = wrow + k0 + 64;
      unsigned short* ad = buf ? adst0 : adst1;
      unsigned short* wd = buf ? wdst0 : wdst1;
      async_copy_b128(an, ad);
      async_copy_b128(an + 8, ad + 8);
      async_copy_b128(wn, wd);
      async_copy_b128(wn + 8, wd + 8);
    }
    const unsigned short* ab = As[buf];
    const unsigned short* wb = Ws[buf];
    for (int kk = 0; kk < 64; kk += 32) {
      v16bf a  = load_frag_a(ab + wr * 16 * 72, 72, lane, kk);
      v16bf b0 = load_frag_b(wb + (wc * 32) * 72, 72, lane, kk);
      v16bf b1 = load_frag_b(wb + (wc * 32 + 16) * 72, 72, lane, kk);
      c0 = wmma_bf16(a, b0, c0);
      c1 = wmma_bf16(a, b1, c1);
    }
    buf ^= 1;
  }

  // ---- epilogue ----
  for (int tile = 0; tile < 2; ++tile) {
    v8f& c = tile ? c1 : c0;
    const int n = n0 + wc * 32 + tile * 16 + lr;
    const float bv = ((flags & 1) && n < N) ? bias[n] : 0.f;
    for (int j = 0; j < 8; ++j) {
      const int m = m0 + wr * 16 + half * 8 + j;
      float val = c[j] + bv;
      if (flags & 2) {  // tanh GELU
        const float x = val;
        const float t = 0.7978845608028654f * (x + 0.044715f * x * x * x);
        val = 0.5f * x * (1.0f + tanhf(t));
      }
      if (n < N) {
        const size_t o = (size_t)m * N + n;
        if (flags & 16) {
          ((unsigned short*)Cout)[o] = f2bf(val);
        } else {
          if (flags & 4) val += Res[o];
          ((float*)Cout)[o] = val;
        }
      }
    }
  }
}

// ---------------------------------------------------------------------------
// Flash-style causal attention on bf16 q/k/v: grid (T/64, NH, B).
// Async-staged Q and K tiles; V transposed into LDS by VALU; scale folded
// into the S spill; online softmax; WMMA for both matmuls.
// ---------------------------------------------------------------------------
__global__ __launch_bounds__(256) void attn_kernel(
    const unsigned short* __restrict__ Qg, const unsigned short* __restrict__ Kg,
    const unsigned short* __restrict__ Vg, unsigned short* __restrict__ Og) {
  __shared__ __align__(16) unsigned short Qs[64 * 72];
  __shared__ __align__(16) unsigned short Ks[64 * 72];
  __shared__ __align__(16) unsigned short Vs[64 * 72];  // transposed [hd][key]
  __shared__ __align__(16) unsigned short Ps[64 * 72];
  __shared__ float Ss[64 * 64];
  __shared__ float rowm[64], rowl[64], rowc[64];

  const int tid  = threadIdx.x;
  const int wave = tid >> 5, lane = tid & 31;
  const int half = (lane >> 4) & 1, lr = lane & 15;
  const int wr = wave & 3, wc = wave >> 2;
  const int qb = blockIdx.x, h = blockIdx.y, b = blockIdx.z;
  const int r  = tid >> 2, cb = (tid & 3) * 16;
  const float scale = 0.125f;  // 1/sqrt(HD=64)

  // stage Q tile asynchronously (consumed after first wait+barrier)
  {
    const unsigned short* qrow =
        Qg + (size_t)(b * GPT_T + qb * 64 + r) * GPT_D + h * GPT_HD + cb;
    async_copy_b128(qrow, &Qs[r * 72 + cb]);
    async_copy_b128(qrow + 8, &Qs[r * 72 + cb + 8]);
  }
  if (tid < 64) { rowm[tid] = -1e30f; rowl[tid] = 0.f; }

  v8f o0 = {}, o1 = {};

  for (int kb = 0; kb <= qb; ++kb) {
    // stage K tile (async) and V tile transposed (VALU)
    {
      const unsigned short* krow =
          Kg + (size_t)(b * GPT_T + kb * 64 + r) * GPT_D + h * GPT_HD + cb;
      async_copy_b128(krow, &Ks[r * 72 + cb]);
      async_copy_b128(krow + 8, &Ks[r * 72 + cb + 8]);
      const unsigned short* vrow =
          Vg + (size_t)(b * GPT_T + kb * 64 + r) * GPT_D + h * GPT_HD;
      for (int g = 0; g < 2; ++g) {
        u32x4 u = *(const u32x4*)(vrow + cb + g * 8);
        const int d0 = cb + g * 8;
        Vs[(d0 + 0) * 72 + r] = (unsigned short)(u.x & 0xffffu);
        Vs[(d0 + 1) * 72 + r] = (unsigned short)(u.x >> 16);
        Vs[(d0 + 2) * 72 + r] = (unsigned short)(u.y & 0xffffu);
        Vs[(d0 + 3) * 72 + r] = (unsigned short)(u.y >> 16);
        Vs[(d0 + 4) * 72 + r] = (unsigned short)(u.z & 0xffffu);
        Vs[(d0 + 5) * 72 + r] = (unsigned short)(u.z >> 16);
        Vs[(d0 + 6) * 72 + r] = (unsigned short)(u.w & 0xffffu);
        Vs[(d0 + 7) * 72 + r] = (unsigned short)(u.w >> 16);
      }
    }
    wait_async0();
    __syncthreads();  // (1) staging visible (covers Q on first iteration)

    // S = Q @ K^T (contraction over hd=64)
    v8f s0 = {}, s1 = {};
    for (int kk = 0; kk < GPT_HD; kk += 32) {
      v16bf a  = load_frag_a(Qs + wr * 16 * 72, 72, lane, kk);
      v16bf b0 = load_frag_b(Ks + (wc * 32) * 72, 72, lane, kk);
      v16bf b1 = load_frag_b(Ks + (wc * 32 + 16) * 72, 72, lane, kk);
      s0 = wmma_bf16(a, b0, s0);
      s1 = wmma_bf16(a, b1, s1);
    }
    // scale + causal mask + spill S to LDS
    for (int tile = 0; tile < 2; ++tile) {
      v8f& s = tile ? s1 : s0;
      const int col   = wc * 32 + tile * 16 + lr;
      const int kglob = kb * 64 + col;
      for (int j = 0; j < 8; ++j) {
        const int row   = wr * 16 + half * 8 + j;
        const int qglob = qb * 64 + row;
        float val = s[j] * scale;
        if (kglob > qglob) val = -1e30f;
        Ss[row * 64 + col] = val;
      }
    }
    __syncthreads();  // (2)

    // online softmax per row
    if (tid < 64) {
      const int row = tid;
      const float mold = rowm[row];
      float mx = mold;
      for (int j = 0; j < 64; ++j) mx = fmaxf(mx, Ss[row * 64 + j]);
      const float corr = __expf(mold - mx);
      float sum = 0.f;
      for (int j = 0; j < 64; ++j) {
        float e = __expf(Ss[row * 64 + j] - mx);
        sum += e;
        Ps[row * 72 + j] = f2bf(e);
      }
      rowc[row] = corr;
      rowm[row] = mx;
      rowl[row] = rowl[row] * corr + sum;
    }
    __syncthreads();  // (3)

    // rescale running O, then O += P @ V
    {
      const int rbase = wr * 16 + half * 8;
      for (int j = 0; j < 8; ++j) {
        const float f = rowc[rbase + j];
        o0[j] *= f;
        o1[j] *= f;
      }
    }
    for (int kk = 0; kk < 64; kk += 32) {
      v16bf a  = load_frag_a(Ps + wr * 16 * 72, 72, lane, kk);
      v16bf b0 = load_frag_b(Vs + (wc * 32) * 72, 72, lane, kk);
      v16bf b1 = load_frag_b(Vs + (wc * 32 + 16) * 72, 72, lane, kk);
      o0 = wmma_bf16(a, b0, o0);
      o1 = wmma_bf16(a, b1, o1);
    }
    __syncthreads();  // (4) protect Ks/Vs/Ps before next iteration restages
  }

  // O /= rowsum, write ctx (bf16)
  for (int tile = 0; tile < 2; ++tile) {
    v8f& o = tile ? o1 : o0;
    const int col = wc * 32 + tile * 16 + lr;
    for (int j = 0; j < 8; ++j) {
      const int row = wr * 16 + half * 8 + j;
      const float val = o[j] / rowl[row];
      Og[(size_t)(b * GPT_T + qb * 64 + row) * GPT_D + h * GPT_HD + col] = f2bf(val);
    }
  }
}

// ---------------------------------------------------------------------------
// Loss: mean NLL with log-softmax over V, one block per row
// ---------------------------------------------------------------------------
__global__ void zero_kernel(float* p) {
  if (threadIdx.x == 0 && blockIdx.x == 0) *p = 0.f;
}

__global__ __launch_bounds__(256) void loss_kernel(
    const float* __restrict__ logits, const int* __restrict__ tgt,
    float* __restrict__ loss) {
  __shared__ float red[256];
  const int row = blockIdx.x;
  const float* l = logits + (size_t)row * GPT_V;
  float mx = -1e30f;
  for (int j = threadIdx.x; j < GPT_V; j += 256) mx = fmaxf(mx, l[j]);
  red[threadIdx.x] = mx;
  __syncthreads();
  for (int st = 128; st > 0; st >>= 1) {
    if (threadIdx.x < st)
      red[threadIdx.x] = fmaxf(red[threadIdx.x], red[threadIdx.x + st]);
    __syncthreads();
  }
  mx = red[0];
  __syncthreads();
  float s = 0.f;
  for (int j = threadIdx.x; j < GPT_V; j += 256) s += __expf(l[j] - mx);
  red[threadIdx.x] = s;
  __syncthreads();
  for (int st = 128; st > 0; st >>= 1) {
    if (threadIdx.x < st) red[threadIdx.x] += red[threadIdx.x + st];
    __syncthreads();
  }
  if (threadIdx.x == 0) {
    const float lse = mx + logf(red[0]);
    const float lp  = l[tgt[row]] - lse;
    atomicAdd(loss, -lp * (1.0f / (float)ROWS));
  }
}

// ---------------------------------------------------------------------------
// Orchestration
// ---------------------------------------------------------------------------
extern "C" void kernel_launch(void* const* d_in, const int* in_sizes, int n_in,
                              void* d_out, int out_size, void* d_ws, size_t ws_size,
                              hipStream_t stream) {
  (void)in_sizes; (void)n_in; (void)out_size; (void)ws_size;

  const float* wte     = (const float*)d_in[0];
  const float* wpe     = (const float*)d_in[1];
  const float* ln_f_g  = (const float*)d_in[2 + 13 * GPT_L + 0];
  const float* ln_f_b  = (const float*)d_in[2 + 13 * GPT_L + 1];
  const int*   idx     = (const int*)  d_in[2 + 13 * GPT_L + 2];
  const int*   targets = (const int*)  d_in[2 + 13 * GPT_L + 3];

  // ---- workspace layout ----
  const size_t ND = (size_t)ROWS * GPT_D;
  char* p = (char*)d_ws;
  float*          x   = (float*)p;          p += ND * 4;
  unsigned short* xn  = (unsigned short*)p; p += ND * 2;
  unsigned short* q   = (unsigned short*)p; p += ND * 2;
  unsigned short* k   = (unsigned short*)p; p += ND * 2;
  unsigned short* v   = (unsigned short*)p; p += ND * 2;
  unsigned short* ctx = (unsigned short*)p; p += ND * 2;
  unsigned short* hfc = (unsigned short*)p; p += (size_t)ROWS * GPT_FF * 2;
  unsigned short* wb  = (unsigned short*)p;  // bf16 weights, [N][K] layouts
  const size_t SQ = (size_t)GPT_D * GPT_D;       // 589,824
  const size_t SF = (size_t)GPT_D * GPT_FF;      // 2,359,296
  const size_t LW = 4 * SQ + 2 * SF;             // per-layer bf16 elems
  unsigned short* wte_b = wb + GPT_L * LW;       // [VP][D]

  float* logits = (float*)d_out;
  float* loss   = logits + (size_t)ROWS * GPT_V;

  const dim3 blk(256);
  const dim3 gD(GPT_D / 64, ROWS / 64);          // (12, 32)
  const dim3 gFF(GPT_FF / 64, ROWS / 64);        // (48, 32)
  const dim3 gV(GPT_VP / 64, ROWS / 64);         // (786, 32)
  const dim3 gAttn(GPT_T / 64, GPT_NH, GPT_B);   // (16, 12, 2)
  const dim3 gTqq(GPT_D / 64, GPT_D / 64);       // (12, 12)
  const dim3 gTfc(GPT_FF / 64, GPT_D / 64);      // (48, 12)
  const dim3 gTpj(GPT_D / 64, GPT_FF / 64);      // (12, 48)

  // ---- per-call weight conversion (memory-bound, ~0.5 GB read) ----
  for (int i = 0; i < GPT_L; ++i) {
    const int base = 2 + 13 * i;
    unsigned short* lw = wb + (size_t)i * LW;
    wconv_t_kernel<<<gTqq, blk, 0, stream>>>((const float*)d_in[base + 2],
                                             lw + 0 * SQ, GPT_D, GPT_D);   // wq
    wconv_t_kernel<<<gTqq, blk, 0, stream>>>((const float*)d_in[base + 3],
                                             lw + 1 * SQ, GPT_D, GPT_D);   // wk
    wconv_t_kernel<<<gTqq, blk, 0, stream>>>((const float*)d_in[base + 4],
                                             lw + 2 * SQ, GPT_D, GPT_D);   // wv
    wconv_t_kernel<<<gTqq, blk, 0, stream>>>((const float*)d_in[base + 5],
                                             lw + 3 * SQ, GPT_D, GPT_D);   // wo
    wconv_t_kernel<<<gTfc, blk, 0, stream>>>((const float*)d_in[base + 9],
                                             lw + 4 * SQ, GPT_D, GPT_FF);  // fc_w -> [FF][D]
    wconv_t_kernel<<<gTpj, blk, 0, stream>>>((const float*)d_in[base + 11],
                                             lw + 4 * SQ + SF, GPT_FF, GPT_D); // proj_w -> [D][FF]
  }
  wconv_pad_kernel<<<GPT_VP, blk, 0, stream>>>(wte, wte_b, GPT_D, GPT_V);

  // ---- forward ----
  embed_kernel<<<ROWS, blk, 0, stream>>>(wte, wpe, idx, x);

  for (int i = 0; i < GPT_L; ++i) {
    const int base = 2 + 13 * i;
    unsigned short* lw = wb + (size_t)i * LW;
    const float* ln1_g  = (const float*)d_in[base + 0];
    const float* ln1_b  = (const float*)d_in[base + 1];
    const float* bo     = (const float*)d_in[base + 6];
    const float* ln2_g  = (const float*)d_in[base + 7];
    const float* ln2_b  = (const float*)d_in[base + 8];
    const float* fc_b   = (const float*)d_in[base + 10];
    const float* proj_b = (const float*)d_in[base + 12];

    ln_kernel<<<ROWS / 8, blk, 0, stream>>>(x, ln1_g, ln1_b, xn, ROWS);
    gemm_bf16_kernel<<<gD, blk, 0, stream>>>(xn, lw + 0 * SQ, nullptr, nullptr,
                                             q, ROWS, GPT_D, GPT_D, 16);
    gemm_bf16_kernel<<<gD, blk, 0, stream>>>(xn, lw + 1 * SQ, nullptr, nullptr,
                                             k, ROWS, GPT_D, GPT_D, 16);
    gemm_bf16_kernel<<<gD, blk, 0, stream>>>(xn, lw + 2 * SQ, nullptr, nullptr,
                                             v, ROWS, GPT_D, GPT_D, 16);
    attn_kernel<<<gAttn, blk, 0, stream>>>(q, k, v, ctx);
    // x = x + ctx @ wo + bo   (fp32 out, residual)
    gemm_bf16_kernel<<<gD, blk, 0, stream>>>(ctx, lw + 3 * SQ, bo, x,
                                             x, ROWS, GPT_D, GPT_D, 1 | 4);
    ln_kernel<<<ROWS / 8, blk, 0, stream>>>(x, ln2_g, ln2_b, xn, ROWS);
    // hfc = gelu(xn @ fc_w + fc_b)  (bf16 out)
    gemm_bf16_kernel<<<gFF, blk, 0, stream>>>(xn, lw + 4 * SQ, fc_b, nullptr,
                                              hfc, ROWS, GPT_FF, GPT_D, 1 | 2 | 16);
    // x = x + hfc @ proj_w + proj_b  (fp32 out, residual)
    gemm_bf16_kernel<<<gD, blk, 0, stream>>>(hfc, lw + 4 * SQ + SF, proj_b, x,
                                             x, ROWS, GPT_D, GPT_FF, 1 | 4);
  }

  ln_kernel<<<ROWS / 8, blk, 0, stream>>>(x, ln_f_g, ln_f_b, xn, ROWS);
  // logits = xn @ wte^T  (wte_b is [VP][D] bf16)
  gemm_bf16_kernel<<<gV, blk, 0, stream>>>(xn, wte_b, nullptr, nullptr,
                                           logits, ROWS, GPT_V, GPT_D, 0);

  zero_kernel<<<1, 1, 0, stream>>>(loss);
  loss_kernel<<<ROWS, blk, 0, stream>>>(logits, targets, loss);
}
